// TransformerEncoderLayerGQA_64338610094879
// MI455X (gfx1250) — compile-verified, matrix operand order checked by reading, not run on hardware
//
#include <hip/hip_runtime.h>
#include <math.h>

// ---------------------------------------------------------------------------
// CDNA5 (gfx1250) wave32 WMMA types & helpers
// ---------------------------------------------------------------------------
typedef _Float16 half_t;
typedef __attribute__((ext_vector_type(16))) _Float16 v16h;
typedef __attribute__((ext_vector_type(8)))  _Float16 v8h;
typedef __attribute__((ext_vector_type(4)))  _Float16 v4h;
typedef __attribute__((ext_vector_type(8)))  float    v8f;

__device__ __forceinline__ v8f wmma_f16(v16h a, v16h b, v8f c) {
  // D = A(16x32 f16) x B(32x16 f16) + C(16x16 f32)
  return __builtin_amdgcn_wmma_f32_16x16x32_f16(
      /*neg_a=*/false, a, /*neg_b=*/false, b,
      /*c_mod=*/(short)0, c, /*reuse_a=*/false, /*reuse_b=*/false);
}

// A/B fragment from LDS (wave32 16-bit layout): lane (l&15) picks row/col,
// g=(l>>4) picks the K sub-group; the 16 halves are two contiguous 8-half
// runs at +8g and +16+8g -> two ds_load_b128 per fragment.
__device__ __forceinline__ v16h frag_lds(const half_t* rowbase, int g) {
  v8h lo = *(const v8h*)(rowbase + 8 * g);
  v8h hi = *(const v8h*)(rowbase + 16 + 8 * g);
  v16h r;
#pragma unroll
  for (int i = 0; i < 8; i++) { r[i] = lo[i]; r[8 + i] = hi[i]; }
  return r;
}

__device__ __forceinline__ float gelu_exact(float x) {
  return 0.5f * x * (1.0f + erff(x * 0.70710678118654752440f));
}

// ---------------------------------------------------------------------------
// fp32 -> f16 pack (O(N^2) conversion hoisted out of the O(N^3) GEMMs)
// ---------------------------------------------------------------------------
__global__ void cvt_f16_kernel(const float* __restrict__ s,
                               half_t* __restrict__ d, size_t n4) {
  size_t i = (size_t)blockIdx.x * blockDim.x + threadIdx.x;
  size_t st = (size_t)gridDim.x * blockDim.x;
  for (; i < n4; i += st) {
    float4 v = reinterpret_cast<const float4*>(s)[i];
    v4h h;
    h[0] = (half_t)v.x; h[1] = (half_t)v.y;
    h[2] = (half_t)v.z; h[3] = (half_t)v.w;
    reinterpret_cast<v4h*>(d)[i] = h;
  }
}

// ---------------------------------------------------------------------------
// Pipelined double-buffered WMMA GEMM, f16 inputs, f32 accumulate.
//   C[b] = act(A[b] @ B[b] + bias) (+ resid)
// Tile 128(M) x 128(N), BK=32, 256 threads = 8 waves; each wave owns a 32x64
// slab (2x4 WMMA tiles -> 64 WMMA/block/k-step). Staging is role-split:
// waves 0-3 stage B (register-transposed b64 stores), waves 4-7 stage A
// (b128 stores); all global loads sit behind persistent incremented pointers
// and are issued before the WMMAs of the previous tile.
// Requires M%128==0, N%128==0, K%32==0 (true at every call site).
// ---------------------------------------------------------------------------
template <typename OutT>
__global__ __launch_bounds__(256) void wmma_gemm_f16(
    const half_t* __restrict__ A, const half_t* __restrict__ B,
    const float* __restrict__ bias, const float* __restrict__ resid,
    OutT* __restrict__ C, int M, int N, int K,
    unsigned long long sA, unsigned long long sB, unsigned long long sC,
    int act /*0=none,1=gelu*/) {
  const int bn = blockIdx.x * 128;
  const int bm = blockIdx.y * 128;
  const int e  = blockIdx.z;
  A += (unsigned long long)e * sA;
  B += (unsigned long long)e * sB;
  C += (unsigned long long)e * sC;

  const int tid  = threadIdx.x;
  const int lane = tid & 31;
  const int w    = tid >> 5;   // wave 0..7
  const int ln   = lane & 15;
  const int g    = lane >> 4;
  const int wr   = w >> 1;     // 0..3 row slab
  const int wc   = w & 1;      // 0..1 col slab

  // 40-half (80B) row stride: keeps 8-half runs 16B aligned, staggers banks.
  __shared__ __align__(16) half_t As[2][128 * 40];
  __shared__ __align__(16) half_t Bs[2][128 * 40];

  const v8f vzero = {0.f, 0.f, 0.f, 0.f, 0.f, 0.f, 0.f, 0.f};
  v8f acc[2][4];
#pragma unroll
  for (int i = 0; i < 2; i++)
#pragma unroll
    for (int j = 0; j < 4; j++) acc[i][j] = vzero;

  // --- role-split staging state (persistent global pointers) ---
  const bool isB = (tid < 128);
  const int kq = (tid >> 4) * 4;          // B micro-tile: 4 k-rows
  const int nq = (tid & 15) * 8;          //               8 n-cols
  v8h sreg[4];
  const half_t* gp[4];
  size_t adv;
  if (isB) {
#pragma unroll
    for (int j = 0; j < 4; j++)
      gp[j] = B + (size_t)(kq + j) * N + bn + nq;
    adv = (size_t)32 * N;
  } else {
#pragma unroll
    for (int i = 0; i < 4; i++) {
      int linear = (tid - 128) + 128 * i;  // 512 v8h cover 128x32 A tile
      int row = linear >> 2;
      int kc  = (linear & 3) * 8;
      gp[i] = A + (size_t)(bm + row) * K + kc;
    }
    adv = 32;
  }

  auto ldg = [&]() {
#pragma unroll
    for (int i = 0; i < 4; i++)
      sreg[i] = *reinterpret_cast<const v8h*>(gp[i]);
#pragma unroll
    for (int i = 0; i < 4; i++) gp[i] += adv;
  };
  auto sts = [&](int buf) {
    if (isB) {
#pragma unroll
      for (int c = 0; c < 8; c++) {        // register-transposed b64 stores
        v4h col;
        col[0] = sreg[0][c]; col[1] = sreg[1][c];
        col[2] = sreg[2][c]; col[3] = sreg[3][c];
        *reinterpret_cast<v4h*>(&Bs[buf][(nq + c) * 40 + kq]) = col;
      }
    } else {
#pragma unroll
      for (int i = 0; i < 4; i++) {
        int linear = (tid - 128) + 128 * i;
        int row = linear >> 2;
        int kc  = (linear & 3) * 8;
        *reinterpret_cast<v8h*>(&As[buf][row * 40 + kc]) = sreg[i];
      }
    }
  };
  auto mma = [&](int buf) {
    v16h af[2], bf[4];
#pragma unroll
    for (int i = 0; i < 2; i++)
      af[i] = frag_lds(&As[buf][(32 * wr + 16 * i + ln) * 40], g);
#pragma unroll
    for (int j = 0; j < 4; j++)
      bf[j] = frag_lds(&Bs[buf][(64 * wc + 16 * j + ln) * 40], g);
#pragma unroll
    for (int i = 0; i < 2; i++)
#pragma unroll
      for (int j = 0; j < 4; j++)
        acc[i][j] = wmma_f16(af[i], bf[j], acc[i][j]);
  };

  const int nkt = K >> 5;
  ldg();
  sts(0);
  for (int kt = 1; kt < nkt; kt++) {
    __syncthreads();
    ldg();                                // issue next-tile loads first
    __builtin_prefetch(gp[0], 0, 3);      // L2 prefetch, tile kt+1
    mma((kt - 1) & 1);                    // WMMAs hide the loads in flight
    sts(kt & 1);                          // waits loads, fills other buffer
  }
  __syncthreads();
  mma((nkt - 1) & 1);

  // epilogue: reg r -> row (r + 8g), col ln
#pragma unroll
  for (int i = 0; i < 2; i++) {
#pragma unroll
    for (int j = 0; j < 4; j++) {
#pragma unroll
      for (int r = 0; r < 8; r++) {
        int row = bm + 32 * wr + 16 * i + r + 8 * g;
        int col = bn + 64 * wc + 16 * j + ln;
        float v = acc[i][j][r];
        if (bias) v += bias[col];
        if (act == 1) v = gelu_exact(v);
        if (resid) v += resid[(size_t)row * N + col];
        C[(size_t)row * N + col] = (OutT)v;
      }
    }
  }
}

// ---------------------------------------------------------------------------
// Flash-attention GQA kernel, all-f16 Q/K/V with f32 online softmax.
// One block = 16 query rows of one (b, h); streams 64-key chunks.
// 128 threads = 4 waves; wave w owns a 16-wide key (scores) / d (output) slab.
// ---------------------------------------------------------------------------
#define HD 64
__global__ __launch_bounds__(128) void gqa_flash_kernel(
    const half_t* __restrict__ q, const half_t* __restrict__ k,
    const half_t* __restrict__ v, half_t* __restrict__ out, int S) {
  const int qb  = blockIdx.x;
  const int h   = blockIdx.y;
  const int b   = blockIdx.z;
  const int kvh = h >> 2;   // NUM_REP = 4

  const int tid  = threadIdx.x;
  const int lane = tid & 31;
  const int w    = tid >> 5;
  const int ln   = lane & 15;
  const int g    = lane >> 4;

  __shared__ __align__(16) half_t qs[16 * HD];
  __shared__ __align__(16) half_t ks[64 * HD];    // [key][d]
  __shared__ __align__(16) half_t vts[HD * 64];   // [d][key]
  __shared__ float  sbuf[16 * 64];
  __shared__ __align__(16) half_t pbuf[16 * 64];
  __shared__ float  red[16 * 8];
  __shared__ float  m_s[16], l_s[16], alpha_s[16];

  const half_t* qg = q + ((size_t)(b * S + qb * 16)) * 1024 + h * HD;
  {
    int row = tid >> 3, cq = (tid & 7) * 8;   // 128 v8h copies
    *reinterpret_cast<v8h*>(&qs[row * HD + cq]) =
        *reinterpret_cast<const v8h*>(qg + (size_t)row * 1024 + cq);
  }
  if (tid < 16) { m_s[tid] = -1e30f; l_s[tid] = 0.f; }
  __syncthreads();

  v16h qa0 = frag_lds(qs + ln * HD + 0,  g);
  v16h qa1 = frag_lds(qs + ln * HD + 32, g);

  const v8f vzero = {0.f, 0.f, 0.f, 0.f, 0.f, 0.f, 0.f, 0.f};
  v8f oacc = vzero;

  // persistent chunk pointers; advance by 64 keys * 256 halves per iteration
  const int keyq = (tid & 15) * 4;   // V micro-tile: 4 keys x 8 d
  const int dq2  = (tid >> 4) * 8;
  const half_t* kPtr = k + ((size_t)(b * S)) * 256 + kvh * HD +
                       (size_t)(tid >> 3) * 256 + (tid & 7) * 8;
  const half_t* vPtr = v + ((size_t)(b * S)) * 256 + kvh * HD +
                       (size_t)keyq * 256 + dq2;
  half_t* ksDst = &ks[(tid >> 3) * HD + (tid & 7) * 8];

  for (int c0 = 0; c0 < S; c0 += 64) {
    __syncthreads();
    // --- stage K/V chunk: batched b128 loads at immediate offsets ---
    v8h kreg[4], vreg[4];
#pragma unroll
    for (int i = 0; i < 4; i++)   // keys stride 16 -> +4096 halves
      kreg[i] = *reinterpret_cast<const v8h*>(kPtr + (size_t)i * 4096);
#pragma unroll
    for (int j = 0; j < 4; j++)
      vreg[j] = *reinterpret_cast<const v8h*>(vPtr + (size_t)j * 256);
    kPtr += (size_t)64 * 256;
    vPtr += (size_t)64 * 256;
#pragma unroll
    for (int i = 0; i < 4; i++)
      *reinterpret_cast<v8h*>(ksDst + i * 16 * HD) = kreg[i];
#pragma unroll
    for (int c = 0; c < 8; c++) {      // register transpose of V
      v4h col;
      col[0] = vreg[0][c]; col[1] = vreg[1][c];
      col[2] = vreg[2][c]; col[3] = vreg[3][c];
      *reinterpret_cast<v4h*>(&vts[(dq2 + c) * 64 + keyq]) = col;
    }
    __syncthreads();

    // --- scores: wave w handles keys [16w, 16w+16) ---
    v8f sc = vzero;
    {
      const half_t* colp = ks + (16 * w + ln) * HD;
      sc = wmma_f16(qa0, frag_lds(colp + 0,  g), sc);
      sc = wmma_f16(qa1, frag_lds(colp + 32, g), sc);
    }
#pragma unroll
    for (int r = 0; r < 8; r++)
      sbuf[(r + 8 * g) * 64 + 16 * w + ln] = sc[r] * 0.125f;  // 1/sqrt(64)
    __syncthreads();

    // --- online softmax (8 threads per row) ---
    {
      int row = tid >> 3, sub = tid & 7;
      float mx = -1e30f;
#pragma unroll
      for (int j = 0; j < 8; j++)
        mx = fmaxf(mx, sbuf[row * 64 + sub * 8 + j]);
      red[row * 8 + sub] = mx;
    }
    __syncthreads();
    if (tid < 16) {
      float mx = red[tid * 8];
#pragma unroll
      for (int j = 1; j < 8; j++) mx = fmaxf(mx, red[tid * 8 + j]);
      float mo = m_s[tid];
      float mn = fmaxf(mo, mx);
      alpha_s[tid] = expf(mo - mn);
      m_s[tid] = mn;
    }
    __syncthreads();
    {
      int row = tid >> 3, sub = tid & 7;
      float mn = m_s[row];
      float sum = 0.f;
#pragma unroll
      for (int j = 0; j < 8; j++) {
        float p = expf(sbuf[row * 64 + sub * 8 + j] - mn);
        pbuf[row * 64 + sub * 8 + j] = (half_t)p;
        sum += p;
      }
      red[row * 8 + sub] = sum;
    }
    __syncthreads();
    if (tid < 16) {
      float s = 0.f;
#pragma unroll
      for (int j = 0; j < 8; j++) s += red[tid * 8 + j];
      l_s[tid] = l_s[tid] * alpha_s[tid] + s;
    }
    __syncthreads();

    // --- rescale running output, then O += P @ V_chunk ---
#pragma unroll
    for (int r = 0; r < 8; r++) oacc[r] *= alpha_s[r + 8 * g];
    {
      const half_t* prow = pbuf + ln * 64;
      const half_t* vcol = vts + (16 * w + ln) * 64;
      oacc = wmma_f16(frag_lds(prow + 0,  g), frag_lds(vcol + 0,  g), oacc);
      oacc = wmma_f16(frag_lds(prow + 32, g), frag_lds(vcol + 32, g), oacc);
    }
  }
  __syncthreads();

  half_t* og = out + ((size_t)(b * S + qb * 16)) * 1024 + h * HD;
#pragma unroll
  for (int r = 0; r < 8; r++) {
    int row = r + 8 * g;
    og[(size_t)row * 1024 + 16 * w + ln] = (half_t)(oacc[r] / l_s[row]);
  }
}

// ---------------------------------------------------------------------------
// RMSNorm over last dim (1024). One block per row.
// ---------------------------------------------------------------------------
__global__ __launch_bounds__(256) void rmsnorm_kernel(
    const float* __restrict__ x, const float* __restrict__ wgt,
    float* __restrict__ y) {
  const int t = blockIdx.x, tid = threadIdx.x;
  __shared__ float red[256];
  float s = 0.f;
  for (int i = tid; i < 1024; i += 256) {
    float v = x[(size_t)t * 1024 + i];
    s += v * v;
  }
  red[tid] = s;
  __syncthreads();
  for (int off = 128; off > 0; off >>= 1) {
    if (tid < off) red[tid] += red[tid + off];
    __syncthreads();
  }
  float inv = rsqrtf(red[0] * (1.0f / 1024.f) + 1e-6f);
  for (int i = tid; i < 1024; i += 256)
    y[(size_t)t * 1024 + i] = wgt[i] * (x[(size_t)t * 1024 + i] * inv);
}

// ---------------------------------------------------------------------------
// MoE gating: sigmoid(x @ gate_w), top-2, normalized weights. Block per token.
// ---------------------------------------------------------------------------
__global__ __launch_bounds__(128) void gate_topk_kernel(
    const float* __restrict__ xn, const float* __restrict__ gw,
    int* __restrict__ sel, float* __restrict__ wts) {
  const int t = blockIdx.x, tid = threadIdx.x;
  __shared__ float red[128 * 8];
  float acc[8] = {0.f, 0.f, 0.f, 0.f, 0.f, 0.f, 0.f, 0.f};
  for (int i = tid; i < 1024; i += 128) {
    float xv = xn[(size_t)t * 1024 + i];
#pragma unroll
    for (int e = 0; e < 8; e++) acc[e] += xv * gw[i * 8 + e];
  }
#pragma unroll
  for (int e = 0; e < 8; e++) red[tid * 8 + e] = acc[e];
  __syncthreads();
  for (int s = 64; s > 0; s >>= 1) {
    if (tid < s)
#pragma unroll
      for (int e = 0; e < 8; e++) red[tid * 8 + e] += red[(tid + s) * 8 + e];
    __syncthreads();
  }
  if (tid == 0) {
    float p[8];
#pragma unroll
    for (int e = 0; e < 8; e++) p[e] = 1.0f / (1.0f + expf(-red[e]));
    int i0 = 0;
    for (int e = 1; e < 8; e++) if (p[e] > p[i0]) i0 = e;
    int i1 = -1;
    for (int e = 0; e < 8; e++)
      if (e != i0 && (i1 < 0 || p[e] > p[i1])) i1 = e;
    float w0 = p[i0], w1 = p[i1], s = w0 + w1 + 1e-6f;
    sel[2 * t + 0] = i0; sel[2 * t + 1] = i1;
    wts[2 * t + 0] = w0 / s; wts[2 * t + 1] = w1 / s;
  }
}

// ---------------------------------------------------------------------------
// Deterministic routing scan (reference cumsum ranks + capacity drop).
// Single block, 256 threads, 8192 entries, Hillis-Steele scan. No atomics.
// ---------------------------------------------------------------------------
__global__ __launch_bounds__(256) void route_scan_kernel(
    const int* __restrict__ sel, int* __restrict__ slot, int capacity) {
  __shared__ int cnt[256 * 8];
  const int tid = threadIdx.x;
  const int base = tid * 32;
  int local[8] = {0, 0, 0, 0, 0, 0, 0, 0};
  for (int i = 0; i < 32; i++) local[sel[base + i]]++;
#pragma unroll
  for (int e = 0; e < 8; e++) cnt[tid * 8 + e] = local[e];
  __syncthreads();
  for (int off = 1; off < 256; off <<= 1) {
    int v[8] = {0, 0, 0, 0, 0, 0, 0, 0};
    if (tid >= off)
#pragma unroll
      for (int e = 0; e < 8; e++) v[e] = cnt[(tid - off) * 8 + e];
    __syncthreads();
    if (tid >= off)
#pragma unroll
      for (int e = 0; e < 8; e++) cnt[tid * 8 + e] += v[e];
    __syncthreads();
  }
  int pref[8];
#pragma unroll
  for (int e = 0; e < 8; e++) pref[e] = cnt[tid * 8 + e] - local[e];
  int run[8] = {0, 0, 0, 0, 0, 0, 0, 0};
  for (int i = 0; i < 32; i++) {
    int e = sel[base + i];
    int rank = pref[e] + run[e];
    run[e]++;
    slot[base + i] = (rank < capacity) ? (e * capacity + rank) : -1;
  }
}

__global__ void zero_kernel(float* __restrict__ p, size_t n) {
  size_t i = (size_t)blockIdx.x * blockDim.x + threadIdx.x;
  size_t stride = (size_t)gridDim.x * blockDim.x;
  for (; i < n; i += stride) p[i] = 0.f;
}

__global__ __launch_bounds__(256) void scatter_kernel(
    const float* __restrict__ xn, const int* __restrict__ slot,
    half_t* __restrict__ padded) {
  const int entry = blockIdx.x;
  const int s = slot[entry];
  if (s < 0) return;
  const int t = entry >> 1;
  for (int i = threadIdx.x; i < 1024; i += 256)
    padded[(size_t)s * 1024 + i] = (half_t)xn[(size_t)t * 1024 + i];
}

__global__ __launch_bounds__(256) void combine_kernel(
    const float* __restrict__ xn, const float* __restrict__ outp,
    const int* __restrict__ slot, const float* __restrict__ wts,
    float* __restrict__ src2) {
  const int t = blockIdx.x;
  const int s0 = slot[2 * t + 0], s1 = slot[2 * t + 1];
  const float w0 = wts[2 * t + 0], w1 = wts[2 * t + 1];
  for (int i = threadIdx.x; i < 1024; i += 256) {
    float a = xn[(size_t)t * 1024 + i];
    if (s0 >= 0) a += w0 * outp[(size_t)s0 * 1024 + i];
    if (s1 >= 0) a += w1 * outp[(size_t)s1 * 1024 + i];
    src2[(size_t)t * 1024 + i] = a;
  }
}

// ---------------------------------------------------------------------------
// Orchestration
// ---------------------------------------------------------------------------
extern "C" void kernel_launch(void* const* d_in, const int* in_sizes, int n_in,
                              void* d_out, int out_size, void* d_ws, size_t ws_size,
                              hipStream_t stream) {
  (void)in_sizes; (void)n_in; (void)out_size; (void)ws_size;
  const float* x       = (const float*)d_in[0];
  const float* qw      = (const float*)d_in[1];
  const float* qb      = (const float*)d_in[2];
  const float* kw      = (const float*)d_in[3];
  const float* kb      = (const float*)d_in[4];
  const float* vw      = (const float*)d_in[5];
  const float* vb      = (const float*)d_in[6];
  const float* ow      = (const float*)d_in[7];
  const float* ob      = (const float*)d_in[8];
  const float* gate_w  = (const float*)d_in[9];
  const float* w1      = (const float*)d_in[10];
  const float* w2      = (const float*)d_in[11];
  const float* norm1_w = (const float*)d_in[12];
  const float* norm2_w = (const float*)d_in[13];

  const int T = 4096, D = 1024, S = 1024, Bsz = 4;
  const int E = 8, CAP = 1280, FF = 2048;

  char* wsp = (char*)d_ws;
  auto alloc = [&](size_t bytes) {
    char* p = wsp;
    wsp += (bytes + 255) & ~(size_t)255;
    return p;
  };
  half_t* x_h    = (half_t*)alloc((size_t)T * D * 2);
  half_t* qw_h   = (half_t*)alloc((size_t)D * D * 2);
  half_t* kw_h   = (half_t*)alloc((size_t)D * 256 * 2);
  half_t* vw_h   = (half_t*)alloc((size_t)D * 256 * 2);
  half_t* ow_h   = (half_t*)alloc((size_t)D * D * 2);
  half_t* w1_h   = (half_t*)alloc((size_t)E * D * FF * 2);
  half_t* w2_h   = (half_t*)alloc((size_t)E * FF * D * 2);
  half_t* qbuf   = (half_t*)alloc((size_t)T * D * 2);
  half_t* kbuf   = (half_t*)alloc((size_t)T * 256 * 2);
  half_t* vbuf   = (half_t*)alloc((size_t)T * 256 * 2);
  half_t* attn   = (half_t*)alloc((size_t)T * D * 2);
  half_t* padded = (half_t*)alloc((size_t)E * CAP * D * 2);
  half_t* hbuf   = (half_t*)alloc((size_t)E * CAP * FF * 2);
  float*  src1   = (float*)alloc((size_t)T * D * 4);
  float*  xn     = (float*)alloc((size_t)T * D * 4);
  float*  src2   = (float*)alloc((size_t)T * D * 4);
  float*  outp   = (float*)alloc((size_t)E * CAP * D * 4);
  float*  wts    = (float*)alloc((size_t)2 * T * 4);
  int*    sel    = (int*)alloc((size_t)2 * T * 4);
  int*    slot   = (int*)alloc((size_t)2 * T * 4);

  // 0) hoisted fp32 -> f16 conversions (O(N^2), amortized vs O(N^3) GEMMs)
  cvt_f16_kernel<<<512, 256, 0, stream>>>(x,  x_h,  (size_t)T * D / 4);
  cvt_f16_kernel<<<512, 256, 0, stream>>>(qw, qw_h, (size_t)D * D / 4);
  cvt_f16_kernel<<<256, 256, 0, stream>>>(kw, kw_h, (size_t)D * 256 / 4);
  cvt_f16_kernel<<<256, 256, 0, stream>>>(vw, vw_h, (size_t)D * 256 / 4);
  cvt_f16_kernel<<<512, 256, 0, stream>>>(ow, ow_h, (size_t)D * D / 4);
  cvt_f16_kernel<<<2048, 256, 0, stream>>>(w1, w1_h, (size_t)E * D * FF / 4);
  cvt_f16_kernel<<<2048, 256, 0, stream>>>(w2, w2_h, (size_t)E * FF * D / 4);

  // 1-3) Q/K/V projections (f16 out, fused bias)
  wmma_gemm_f16<half_t><<<dim3(D / 128, T / 128, 1), 256, 0, stream>>>(
      x_h, qw_h, qb, nullptr, qbuf, T, D, D, 0, 0, 0, 0);
  wmma_gemm_f16<half_t><<<dim3(256 / 128, T / 128, 1), 256, 0, stream>>>(
      x_h, kw_h, kb, nullptr, kbuf, T, 256, D, 0, 0, 0, 0);
  wmma_gemm_f16<half_t><<<dim3(256 / 128, T / 128, 1), 256, 0, stream>>>(
      x_h, vw_h, vb, nullptr, vbuf, T, 256, D, 0, 0, 0, 0);

  // 4) flash GQA attention (f16 in/out)
  gqa_flash_kernel<<<dim3(S / 16, 16, Bsz), 128, 0, stream>>>(
      qbuf, kbuf, vbuf, attn, S);

  // 5) output projection + residual: src1 = x + attn @ ow + ob  (f32 out)
  wmma_gemm_f16<float><<<dim3(D / 128, T / 128, 1), 256, 0, stream>>>(
      attn, ow_h, ob, x, src1, T, D, D, 0, 0, 0, 0);

  // 6) RMSNorm 1
  rmsnorm_kernel<<<T, 256, 0, stream>>>(src1, norm1_w, xn);

  // 7-9) MoE routing (deterministic)
  gate_topk_kernel<<<T, 128, 0, stream>>>(xn, gate_w, sel, wts);
  route_scan_kernel<<<1, 256, 0, stream>>>(sel, slot, CAP);
  zero_kernel<<<1024, 256, 0, stream>>>((float*)padded, (size_t)E * CAP * D / 2);
  scatter_kernel<<<2 * T, 256, 0, stream>>>(xn, slot, padded);

  // 10) hbuf = gelu(padded @ w1[e])  (batched, f16 out)
  wmma_gemm_f16<half_t><<<dim3(FF / 128, CAP / 128, E), 256, 0, stream>>>(
      padded, w1_h, nullptr, nullptr, hbuf, CAP, FF, D,
      (unsigned long long)CAP * D, (unsigned long long)D * FF,
      (unsigned long long)CAP * FF, 1);

  // 11) outp = hbuf @ w2[e]  (batched, f32 out)
  wmma_gemm_f16<float><<<dim3(D / 128, CAP / 128, E), 256, 0, stream>>>(
      hbuf, w2_h, nullptr, nullptr, outp, CAP, D, FF,
      (unsigned long long)CAP * FF, (unsigned long long)FF * D,
      (unsigned long long)CAP * D, 0);

  // 12) combine experts + residual
  combine_kernel<<<T, 256, 0, stream>>>(xn, outp, slot, wts, src2);

  // 13) final RMSNorm -> d_out
  rmsnorm_kernel<<<T, 256, 0, stream>>>(src2, norm2_w, (float*)d_out);
}
